// GCN_33165737459841
// MI455X (gfx1250) — compile-verified, matrix operand order checked by reading, not run on hardware
//
#include <hip/hip_runtime.h>
#include <math.h>

typedef __attribute__((ext_vector_type(16))) _Float16 v16h;
typedef __attribute__((ext_vector_type(8)))  float    v8f;

#define NFEAT 128

// ---------------- init: deg=1, agg4=0, agg2=0 ----------------
__global__ void k_init(float* __restrict__ deg, float* __restrict__ agg4,
                       float* __restrict__ agg2, int n) {
    int i = blockIdx.x * blockDim.x + threadIdx.x;
    if (i < 4 * n) agg4[i] = 0.0f;
    if (i < 2 * n) agg2[i] = 0.0f;
    if (i < n)     deg[i]  = 1.0f;
}

// ---------------- degree accumulate over edges ----------------
__global__ void k_deg(const long long* __restrict__ dst,
                      float* __restrict__ deg, int e_count) {
    int e = blockIdx.x * blockDim.x + threadIdx.x;
    if (e >= e_count) return;
    int d = (int)dst[e];
    unsafeAtomicAdd(&deg[d], 1.0f);
}

__global__ void k_rsqrt(float* __restrict__ deg_dinv, int n) {
    int i = blockIdx.x * blockDim.x + threadIdx.x;
    if (i < n) deg_dinv[i] = rsqrtf(deg_dinv[i]);
}

// ---------------- t1 = x @ W1 via WMMA f16 (K=128) ----------------
// One wave computes a 16x16 output tile (cols 0..3 valid). N%16==0 so no tail.
// W1 is staged once per block into LDS as a zero-padded, transposed f16 tile
// Wt[n][k], so each lane's B fragment is two contiguous 16B LDS reads.
__global__ void k_xform1_wmma(const float* __restrict__ x,
                              const float* __restrict__ W1,
                              float* __restrict__ t1, int n_tiles) {
    __shared__ _Float16 Wt[16][NFEAT];        // 4KB: Wt[n][k] = W1[k][n] (n<4), else 0

    int tid = (int)threadIdx.x;
    for (int idx = tid; idx < 16 * NFEAT; idx += (int)blockDim.x) {
        int nn = idx >> 7;                    // 0..15 (column of padded B)
        int k  = idx & (NFEAT - 1);           // 0..127
        Wt[nn][k] = (_Float16)((nn < 4) ? W1[(size_t)k * 4 + nn] : 0.0f);
    }
    __syncthreads();

    int wid  = tid >> 5;
    int tile = blockIdx.x * (int)(blockDim.x >> 5) + wid;
    if (tile >= n_tiles) return;              // wave-uniform: EXEC all-1 for WMMA
    int lane = tid & 31;
    int lo16 = lane & 15;
    int hi   = lane >> 4;                     // 0: K 0-7/16-23, 1: K 8-15/24-31
    int m    = tile * 16 + lo16;              // A-matrix row held by this lane
    int nn   = lo16;                          // B/D column held by this lane

    v8f c = {};
#pragma unroll
    for (int ko = 0; ko < NFEAT; ko += 32) {
        int kb = ko + hi * 8;
        v16h a, b;
        const float* xr = x + (size_t)m * NFEAT + kb;
#pragma unroll
        for (int j = 0; j < 8; ++j) {
            a[j]     = (_Float16)xr[j];
            a[j + 8] = (_Float16)xr[j + 16];
        }
        const _Float16* wr = &Wt[nn][kb];     // contiguous halves -> ds_load_b128
#pragma unroll
        for (int j = 0; j < 8; ++j) {
            b[j]     = wr[j];
            b[j + 8] = wr[j + 16];
        }
        c = __builtin_amdgcn_wmma_f32_16x16x32_f16(false, a, false, b,
                                                   (short)0, c, false, false);
    }
    if (nn < 4) {
#pragma unroll
        for (int r = 0; r < 8; ++r) {
            int row = tile * 16 + hi * 8 + r;     // C/D layout: VGPR r -> M=r (+8 for hi lanes)
            t1[(size_t)row * 4 + nn] = c[r];
        }
    }
}

// ---------------- edge aggregation, 4 components ----------------
__global__ void k_edge4(const long long* __restrict__ src,
                        const long long* __restrict__ dst,
                        const float* __restrict__ dinv,
                        const float* __restrict__ t,
                        float* __restrict__ agg, int e_count) {
    int e = blockIdx.x * blockDim.x + threadIdx.x;
    if (e >= e_count) return;
    int s = (int)src[e];
    int d = (int)dst[e];
    float w = dinv[s] * dinv[d];
    const float4 h = *(const float4*)(t + (size_t)s * 4);
    float* ap = agg + (size_t)d * 4;
    unsafeAtomicAdd(ap + 0, h.x * w);
    unsafeAtomicAdd(ap + 1, h.y * w);
    unsafeAtomicAdd(ap + 2, h.z * w);
    unsafeAtomicAdd(ap + 3, h.w * w);
}

// ---------------- edge aggregation, 2 components ----------------
__global__ void k_edge2(const long long* __restrict__ src,
                        const long long* __restrict__ dst,
                        const float* __restrict__ dinv,
                        const float* __restrict__ t,
                        float* __restrict__ agg, int e_count) {
    int e = blockIdx.x * blockDim.x + threadIdx.x;
    if (e >= e_count) return;
    int s = (int)src[e];
    int d = (int)dst[e];
    float w = dinv[s] * dinv[d];
    const float2 h = *(const float2*)(t + (size_t)s * 2);
    float* ap = agg + (size_t)d * 2;
    unsafeAtomicAdd(ap + 0, h.x * w);
    unsafeAtomicAdd(ap + 1, h.y * w);
}

// -------- finalize layer (4 comps): h=tanh(agg+t*self+b), t_next=h@Wn (4x4),
//          write t_next over t, zero agg for next layer --------
__global__ void k_fin4to4(float* __restrict__ t, float* __restrict__ agg,
                          const float* __restrict__ dinv,
                          const float* __restrict__ bvec,
                          const float* __restrict__ Wn, int n) {
    int i = blockIdx.x * blockDim.x + threadIdx.x;
    if (i >= n) return;
    float s  = dinv[i] * dinv[i];
    float4 tv = *(float4*)(t   + (size_t)i * 4);
    float4 av = *(float4*)(agg + (size_t)i * 4);
    float h0 = tanhf(av.x + tv.x * s + bvec[0]);
    float h1 = tanhf(av.y + tv.y * s + bvec[1]);
    float h2 = tanhf(av.z + tv.z * s + bvec[2]);
    float h3 = tanhf(av.w + tv.w * s + bvec[3]);
    float4 o;
    o.x = h0 * Wn[0] + h1 * Wn[4] + h2 * Wn[8]  + h3 * Wn[12];
    o.y = h0 * Wn[1] + h1 * Wn[5] + h2 * Wn[9]  + h3 * Wn[13];
    o.z = h0 * Wn[2] + h1 * Wn[6] + h2 * Wn[10] + h3 * Wn[14];
    o.w = h0 * Wn[3] + h1 * Wn[7] + h2 * Wn[11] + h3 * Wn[15];
    *(float4*)(t   + (size_t)i * 4) = o;
    *(float4*)(agg + (size_t)i * 4) = make_float4(0.f, 0.f, 0.f, 0.f);
}

// -------- finalize layer 2 (4 comps -> 2): t3 = h @ W3 (4x2) --------
__global__ void k_fin4to2(const float* __restrict__ t, const float* __restrict__ agg,
                          const float* __restrict__ dinv,
                          const float* __restrict__ bvec,
                          const float* __restrict__ W3,
                          float* __restrict__ t3, int n) {
    int i = blockIdx.x * blockDim.x + threadIdx.x;
    if (i >= n) return;
    float s  = dinv[i] * dinv[i];
    float4 tv = *(const float4*)(t   + (size_t)i * 4);
    float4 av = *(const float4*)(agg + (size_t)i * 4);
    float h0 = tanhf(av.x + tv.x * s + bvec[0]);
    float h1 = tanhf(av.y + tv.y * s + bvec[1]);
    float h2 = tanhf(av.z + tv.z * s + bvec[2]);
    float h3 = tanhf(av.w + tv.w * s + bvec[3]);
    float2 o;
    o.x = h0 * W3[0] + h1 * W3[2] + h2 * W3[4] + h3 * W3[6];
    o.y = h0 * W3[1] + h1 * W3[3] + h2 * W3[5] + h3 * W3[7];
    *(float2*)(t3 + (size_t)i * 2) = o;
}

// -------- finalize layer 3 + classifier: write h [N,2] and out [N,7] --------
__global__ void k_fin_out(const float* __restrict__ t3, const float* __restrict__ agg3,
                          const float* __restrict__ dinv,
                          const float* __restrict__ b3,
                          const float* __restrict__ Wc,
                          const float* __restrict__ bc,
                          float* __restrict__ out7,    // d_out + 0
                          float* __restrict__ hout,    // d_out + 7*n
                          int n) {
    int i = blockIdx.x * blockDim.x + threadIdx.x;
    if (i >= n) return;
    float s  = dinv[i] * dinv[i];
    float2 tv = *(const float2*)(t3   + (size_t)i * 2);
    float2 av = *(const float2*)(agg3 + (size_t)i * 2);
    float h0 = tanhf(av.x + tv.x * s + b3[0]);
    float h1 = tanhf(av.y + tv.y * s + b3[1]);
    hout[(size_t)i * 2 + 0] = h0;
    hout[(size_t)i * 2 + 1] = h1;
    float* op = out7 + (size_t)i * 7;
#pragma unroll
    for (int c = 0; c < 7; ++c)
        op[c] = h0 * Wc[c] + h1 * Wc[7 + c] + bc[c];
}

extern "C" void kernel_launch(void* const* d_in, const int* in_sizes, int n_in,
                              void* d_out, int out_size, void* d_ws, size_t ws_size,
                              hipStream_t stream) {
    const float*      x    = (const float*)d_in[0];
    const long long*  ei   = (const long long*)d_in[1];
    const float*      W1   = (const float*)d_in[2];
    const float*      b1   = (const float*)d_in[3];
    const float*      W2   = (const float*)d_in[4];
    const float*      b2   = (const float*)d_in[5];
    const float*      W3   = (const float*)d_in[6];
    const float*      b3   = (const float*)d_in[7];
    const float*      Wc   = (const float*)d_in[8];
    const float*      bc   = (const float*)d_in[9];

    const int n = in_sizes[0] / NFEAT;       // 100000
    const int e = in_sizes[1] / 2;           // 3200000
    const long long* src = ei;
    const long long* dst = ei + e;

    // workspace layout (floats): dinv[n] | bufA[4n] | bufB[4n] | t3[2n] | agg3[2n]
    float* ws   = (float*)d_ws;
    float* dinv = ws;
    float* bufA = ws + (size_t)n;
    float* bufB = ws + (size_t)5 * n;
    float* t3   = ws + (size_t)9 * n;
    float* agg3 = ws + (size_t)11 * n;

    float* out7 = (float*)d_out;
    float* hout = (float*)d_out + (size_t)7 * n;

    const int B = 256;
    const int gN  = (n + B - 1) / B;
    const int g4N = (4 * n + B - 1) / B;
    const int gE  = (e + B - 1) / B;
    const int n_tiles = n / 16;                       // 6250, exact
    const int gT  = (n_tiles + (B / 32) - 1) / (B / 32);

    // degrees + init (re-run every call -> deterministic under graph replay)
    k_init  <<<g4N, B, 0, stream>>>(dinv, bufB, agg3, n);
    k_deg   <<<gE,  B, 0, stream>>>(dst, dinv, e);
    k_rsqrt <<<gN,  B, 0, stream>>>(dinv, n);

    // layer 1: WMMA transform, aggregate, finalize(+W2)
    k_xform1_wmma<<<gT, B, 0, stream>>>(x, W1, bufA, n_tiles);
    k_edge4 <<<gE, B, 0, stream>>>(src, dst, dinv, bufA, bufB, e);
    k_fin4to4<<<gN, B, 0, stream>>>(bufA, bufB, dinv, b1, W2, n);

    // layer 2: aggregate, finalize(+W3 -> 2 comps)
    k_edge4 <<<gE, B, 0, stream>>>(src, dst, dinv, bufA, bufB, e);
    k_fin4to2<<<gN, B, 0, stream>>>(bufA, bufB, dinv, b2, W3, t3, n);

    // layer 3: aggregate (2 comps), finalize + classifier
    k_edge2 <<<gE, B, 0, stream>>>(src, dst, dinv, t3, agg3, e);
    k_fin_out<<<gN, B, 0, stream>>>(t3, agg3, dinv, b3, Wc, bc, out7, hout, n);
}
